// SelfAttentionBlock_11879879541687
// MI455X (gfx1250) — matrix operand order repaired and checked by reading, not compiled, not run on hardware
//
#include <hip/hip_runtime.h>
#include <stdint.h>

// ---------------------------------------------------------------------------
// SelfAttentionBlock for MI455X (gfx1250, wave32, WMMA f16 -> f32 accumulate)
// B=4, C=256, H=W=64 -> N=4096, C8=32
// Flash-attention style; LDS tiles filled with global_load_async_to_lds_b128
// (ASYNCcnt) and double buffering; m-tile = 64 to amortize softmax VALU.
// ---------------------------------------------------------------------------

#define B_   4
#define C_   256
#define C8_  32
#define N_   4096

typedef __attribute__((ext_vector_type(16))) _Float16 v16h;
typedef __attribute__((ext_vector_type(8)))  _Float16 v8h;
typedef __attribute__((ext_vector_type(4)))  _Float16 v4h;
typedef __attribute__((ext_vector_type(8)))  float    v8f;

static __device__ __forceinline__ v16h concat8(v8h a, v8h b) {
  return __builtin_shufflevector(a, b, 0,1,2,3,4,5,6,7,8,9,10,11,12,13,14,15);
}

static __device__ __forceinline__ v8f wmma_f16(v16h a, v16h b, v8f c) {
  return __builtin_amdgcn_wmma_f32_16x16x32_f16(false, a, false, b, (short)0, c,
                                                false, false);
}

// Async DMA: 16B per lane, global -> LDS, tracked by ASYNCcnt.
// LDS address = low 32 bits of the generic pointer.
static __device__ __forceinline__ void async_ld_b128(void* lds_dst,
                                                     const void* gsrc) {
  asm volatile("global_load_async_to_lds_b128 %0, %1, off"
               :: "v"((uint32_t)(uintptr_t)lds_dst),
                  "v"((uint64_t)(uintptr_t)gsrc)
               : "memory");
}
#define WAIT_ASYNC(n) asm volatile("s_wait_asynccnt " #n ::: "memory")

// row reductions across the 16 lanes of a half-wave
static __device__ __forceinline__ float redmax16(float v) {
  v = fmaxf(v, __shfl_xor(v, 1, 32));
  v = fmaxf(v, __shfl_xor(v, 2, 32));
  v = fmaxf(v, __shfl_xor(v, 4, 32));
  v = fmaxf(v, __shfl_xor(v, 8, 32));
  return v;
}
static __device__ __forceinline__ float redsum16(float v) {
  v += __shfl_xor(v, 1, 32);
  v += __shfl_xor(v, 2, 32);
  v += __shfl_xor(v, 4, 32);
  v += __shfl_xor(v, 8, 32);
  return v;
}

// ---------------------------------------------------------------------------
// Phase 0a: x [B][C][N] f32 -> xh [B][N][C] f16 (32x32 tile transpose via LDS)
// ---------------------------------------------------------------------------
__global__ __launch_bounds__(256) void sa_xpose_kernel(
    const float* __restrict__ x, _Float16* __restrict__ xh) {
  __shared__ _Float16 tile[32 * 33];
  int bid   = blockIdx.x;
  int ntile = bid & 127;
  int ctile = (bid >> 7) & 7;
  int b     = bid >> 10;
  int n0 = ntile * 32, c0 = ctile * 32;
  int t  = threadIdx.x;

  int cl = t >> 3;
  int nl = (t & 7) << 2;
  float4 v = *(const float4*)(x + (((size_t)b * C_ + c0 + cl) * N_ + n0 + nl));
  tile[cl * 33 + nl + 0] = (_Float16)v.x;
  tile[cl * 33 + nl + 1] = (_Float16)v.y;
  tile[cl * 33 + nl + 2] = (_Float16)v.z;
  tile[cl * 33 + nl + 3] = (_Float16)v.w;
  __syncthreads();

  int nr = t >> 3;
  int cr = (t & 7) << 2;
  v4h o;
  o[0] = tile[(cr + 0) * 33 + nr];
  o[1] = tile[(cr + 1) * 33 + nr];
  o[2] = tile[(cr + 2) * 33 + nr];
  o[3] = tile[(cr + 3) * 33 + nr];
  *(v4h*)(xh + (((size_t)b * N_ + n0 + nr) * C_ + c0 + cr)) = o;
}

// ---------------------------------------------------------------------------
// Phase 0b: concat Wq|Wk|Wv -> Wh [320][256] f16
// ---------------------------------------------------------------------------
__global__ __launch_bounds__(256) void sa_wconv_kernel(
    const float* __restrict__ Wq, const float* __restrict__ Wk,
    const float* __restrict__ Wv, _Float16* __restrict__ Wh) {
  int o = blockIdx.x;   // 0..319
  int c = threadIdx.x;  // 0..255
  float v;
  if (o < 32)       v = Wq[o * C_ + c];
  else if (o < 64)  v = Wk[(o - 32) * C_ + c];
  else              v = Wv[(o - 64) * C_ + c];
  Wh[(size_t)o * C_ + c] = (_Float16)v;
}

// ---------------------------------------------------------------------------
// Phase 1: projections D[n][o] = xh[n][:] . Wh[o][:] + bias.
// grid = B*(N/128) = 128 WGs x 256 thr. Wave = 16 n-rows x 20 o-tiles.
// Weights streamed through double-buffered LDS via async loads; A/B fragments
// software-pipelined one step ahead of their WMMA.
// ---------------------------------------------------------------------------
__global__ __launch_bounds__(256) void sa_proj_kernel(
    const _Float16* __restrict__ xh, const _Float16* __restrict__ Wh,
    const float* __restrict__ bq, const float* __restrict__ bk,
    const float* __restrict__ bv,
    _Float16* __restrict__ qh, _Float16* __restrict__ kh,
    _Float16* __restrict__ vh) {
  __shared__ _Float16 wlds[2][320 * 40];  // [o][ck-chunk 32], stride 40 halves
  __shared__ _Float16 vstage[16 * 136];

  int wgid = blockIdx.x;
  int b  = wgid >> 5;
  int n0 = (wgid & 31) * 128;
  int t  = threadIdx.x;
  int w    = t >> 5;
  int lane = t & 31;
  int l15  = lane & 15;
  int hf   = lane >> 4;

  v8f acc[20];
#pragma unroll
  for (int i = 0; i < 20; i++) acc[i] = v8f{};

  // Wh chunk fill: 320 rows x 32 halves = 1280 b128 chunks, 5 per thread.
  auto issue_fill = [&](int buf, int ck) {
#pragma unroll
    for (int i = 0; i < 5; i++) {
      int cidx = t * 5 + i;
      int row = cidx >> 2, ch = cidx & 3;
      async_ld_b128(&wlds[buf][row * 40 + ch * 8],
                    Wh + (size_t)row * C_ + ck + ch * 8);
    }
  };

  const _Float16* arow = xh + ((size_t)b * N_ + n0 + w * 16 + l15) * C_;
  auto load_af = [&](int ck) -> v16h {
    return concat8(*(const v8h*)(arow + ck + 8 * hf),
                   *(const v8h*)(arow + ck + 16 + 8 * hf));
  };
  auto load_bf = [&](const _Float16* wb, int ot) -> v16h {
    const _Float16* brow = wb + (ot * 16 + l15) * 40 + 8 * hf;
    return concat8(*(const v8h*)brow, *(const v8h*)(brow + 16));
  };

  issue_fill(0, 0);
  v16h af = load_af(0);
  for (int kk = 0; kk < 8; kk++) {
    if (kk < 7) { issue_fill((kk + 1) & 1, (kk + 1) * 32); WAIT_ASYNC(5); }
    else        { WAIT_ASYNC(0); }
    __syncthreads();

    const _Float16* wb = &wlds[kk & 1][0];
    v16h afn = af;
    if (kk < 7) afn = load_af((kk + 1) * 32);  // prefetch next A fragment
    v16h bf = load_bf(wb, 0);
#pragma unroll
    for (int ot = 0; ot < 20; ot++) {
      v16h bfn = bf;
      if (ot < 19) bfn = load_bf(wb, ot + 1);  // prefetch next B fragment
      acc[ot] = wmma_f16(af, bf, acc[ot]);
      bf = bfn;
    }
    af = afn;
    __syncthreads();  // all waves done with this buffer before it is refilled
  }

  // --- q/k epilogue: D row n = g + 8*hf (+wave base), col o = l15 ---
#pragma unroll
  for (int ot = 0; ot < 4; ot++) {
    int o = ot * 16 + l15;
    float bias;
    _Float16* dst;
    int oo;
    if (ot < 2) { bias = bq[o];      dst = qh; oo = o; }
    else        { bias = bk[o - 32]; dst = kh; oo = o - 32; }
#pragma unroll
    for (int g = 0; g < 8; g++) {
      int n = n0 + w * 16 + g + 8 * hf;
      dst[((size_t)b * N_ + n) * C8_ + oo] = (_Float16)(acc[ot][g] + bias);
    }
  }

  // --- v epilogue via LDS so global stores are [c][n]-coalesced ---
#pragma unroll
  for (int ot = 4; ot < 20; ot++) {
    float bias = bv[ot * 16 + l15 - 64];
#pragma unroll
    for (int g = 0; g < 8; g++) {
      int nl = w * 16 + g + 8 * hf;
      vstage[l15 * 136 + nl] = (_Float16)(acc[ot][g] + bias);
    }
    __syncthreads();
    int orow  = t >> 4;
    int chunk = t & 15;
    int c = ot * 16 + orow - 64;
    v8h pay = *(const v8h*)(&vstage[orow * 136 + chunk * 8]);
    *(v8h*)(vh + (((size_t)b * C_ + c) * N_ + n0 + chunk * 8)) = pay;
    __syncthreads();
  }
}

// ---------------------------------------------------------------------------
// Phase 2: flash attention + epilogue (gamma*out + x).
// grid = B*(N/128) = 128 WGs x 8 waves. Each wave owns 16 queries x all 256 c.
// m-tile = 64: S = QK^T (4 WMMA), private online softmax (P/r/l in wave-private
// LDS; same-wave LDS is in-order so no barrier), O += V x P (32 WMMA).
// K/V tiles double-buffered, filled with async b128 DMA (exactly 9 ops/thread).
// V fragments software-pipelined one tile ahead.
// ---------------------------------------------------------------------------
__global__ __launch_bounds__(256) void sa_attn_kernel(
    const _Float16* __restrict__ qh, const _Float16* __restrict__ kh,
    const _Float16* __restrict__ vh, const float* __restrict__ x,
    const float* __restrict__ gamma, float* __restrict__ out) {
  __shared__ _Float16 klds[2][64 * 40];    // [m][c8], stride 40 halves
  __shared__ _Float16 vlds[2][256 * 72];   // [c][m],  stride 72 halves
  __shared__ _Float16 plds[8][16 * 72];    // per-wave [n][m]
  __shared__ float    rl[8][16];
  __shared__ float    ll[8][16];

  int wgid = blockIdx.x;       // B * 32
  int b  = wgid >> 5;
  int n0 = (wgid & 31) * 128;
  int t  = threadIdx.x;
  int w    = t >> 5;
  int lane = t & 31;
  int l15  = lane & 15;
  int hf   = lane >> 4;
  int nq = n0 + w * 16;        // this wave's query base

  v8f acc[16];
#pragma unroll
  for (int i = 0; i < 16; i++) acc[i] = v8f{};

  float mst[8], lst[8];
#pragma unroll
  for (int g = 0; g < 8; g++) { mst[g] = -1e30f; lst[g] = 0.0f; }

  const _Float16* qrow = qh + ((size_t)b * N_ + nq + l15) * C8_;
  v16h qa = concat8(*(const v8h*)(qrow + 8 * hf),
                    *(const v8h*)(qrow + 16 + 8 * hf));

  // K tile: 64x32h = 256 chunks; V tile: 256x64h = 2048 chunks.
  // 2304 chunks / 256 threads = exactly 9 async ops per thread (uniform).
  auto issue_fill = [&](int buf, int m0) {
#pragma unroll
    for (int i = 0; i < 9; i++) {
      int cidx = t * 9 + i;
      if (cidx < 256) {
        int row = cidx >> 2, ch = cidx & 3;
        async_ld_b128(&klds[buf][row * 40 + ch * 8],
                      kh + ((size_t)b * N_ + m0 + row) * C8_ + ch * 8);
      } else {
        int vi = cidx - 256;
        int row = vi >> 3, ch = vi & 7;
        async_ld_b128(&vlds[buf][row * 72 + ch * 8],
                      vh + ((size_t)b * C_ + row) * N_ + m0 + ch * 8);
      }
    }
  };

  issue_fill(0, 0);
  for (int step = 0; step < 64; step++) {
    int cur = step & 1;
    if (step < 63) { issue_fill(cur ^ 1, (step + 1) * 64); WAIT_ASYNC(9); }
    else           { WAIT_ASYNC(0); }
    __syncthreads();

    const _Float16* kb_ = &klds[cur][0];
    const _Float16* vb_ = &vlds[cur][0];

    // S = Q (16x32) x K^T (32x64) as four 16x16 WMMAs
    v8f z = v8f{};
    v8f s[4];
#pragma unroll
    for (int ms = 0; ms < 4; ms++) {
      const _Float16* kr = kb_ + (ms * 16 + l15) * 40 + 8 * hf;
      v16h kb = concat8(*(const v8h*)kr, *(const v8h*)(kr + 16));
      s[ms] = wmma_f16(qa, kb, z);
    }

    // private online softmax over 64 columns; row = g + 8*hf
    _Float16* pw = &plds[w][0];
#pragma unroll
    for (int g = 0; g < 8; g++) {
      float mx = fmaxf(fmaxf(s[0][g], s[1][g]), fmaxf(s[2][g], s[3][g]));
      mx = redmax16(mx);
      float mn = fmaxf(mst[g], mx);
      float r  = __expf(mst[g] - mn);
      float p0 = __expf(s[0][g] - mn);
      float p1 = __expf(s[1][g] - mn);
      float p2 = __expf(s[2][g] - mn);
      float p3 = __expf(s[3][g] - mn);
      float sum = redsum16((p0 + p1) + (p2 + p3));
      lst[g] = lst[g] * r + sum;
      mst[g] = mn;
      int nrow = g + 8 * hf;
      pw[nrow * 72 + l15]      = (_Float16)p0;
      pw[nrow * 72 + 16 + l15] = (_Float16)p1;
      pw[nrow * 72 + 32 + l15] = (_Float16)p2;
      pw[nrow * 72 + 48 + l15] = (_Float16)p3;
      if (l15 == g) rl[w][nrow] = r;  // identical across the half's lanes
    }

    // same-wave LDS is in-order: safe to read our own P/r without a barrier
    float rr = rl[w][l15];  // rescale factor for this lane's query column
    const _Float16* pr = pw + l15 * 72 + 8 * hf;
    v16h pb0 = concat8(*(const v8h*)pr,        *(const v8h*)(pr + 16));
    v16h pb1 = concat8(*(const v8h*)(pr + 32), *(const v8h*)(pr + 48));

    auto load_va = [&](int tile, int koff) -> v16h {
      const _Float16* vr = vb_ + (tile * 16 + l15) * 72 + koff;
      return concat8(*(const v8h*)(vr + 8 * hf),
                     *(const v8h*)(vr + 16 + 8 * hf));
    };

    v16h va0 = load_va(0, 0);
    v16h va1 = load_va(0, 32);
#pragma unroll
    for (int tile = 0; tile < 16; tile++) {
      v16h na0 = va0, na1 = va1;
      if (tile < 15) {                 // prefetch next tile's V fragments
        na0 = load_va(tile + 1, 0);
        na1 = load_va(tile + 1, 32);
      }
#pragma unroll
      for (int g = 0; g < 8; g++) acc[tile][g] *= rr;
      acc[tile] = wmma_f16(va0, pb0, acc[tile]);
      acc[tile] = wmma_f16(va1, pb1, acc[tile]);
      va0 = na0;
      va1 = na1;
    }
    __syncthreads();  // all waves done with this buffer before it is refilled
  }

  // epilogue: out = gamma * (O / l) + x ; stores coalesced over n (= l15)
#pragma unroll
  for (int g = 0; g < 8; g++) {
    if (l15 == g) ll[w][g + 8 * hf] = lst[g];
  }
  float linv = 1.0f / ll[w][l15];
  float gm = gamma[0];
#pragma unroll
  for (int tile = 0; tile < 16; tile++) {
#pragma unroll
    for (int g = 0; g < 8; g++) {
      int c = tile * 16 + g + 8 * hf;
      int n = nq + l15;
      size_t idx = ((size_t)b * C_ + c) * N_ + n;
      out[idx] = gm * (acc[tile][g] * linv) + x[idx];
    }
  }
}

// ---------------------------------------------------------------------------
// Host launcher. Workspace (~18.2 MB):
//   xh [4][4096][256] @ 0          (8,388,608 B)
//   Wh [320][256]     @ 8,388,608  (  163,840 B)
//   qh [4][4096][32]  @ 8,552,448  (1,048,576 B)
//   kh [4][4096][32]  @ 9,601,024  (1,048,576 B)
//   vh [4][256][4096] @ 10,649,600 (8,388,608 B)
// ---------------------------------------------------------------------------
extern "C" void kernel_launch(void* const* d_in, const int* in_sizes, int n_in,
                              void* d_out, int out_size, void* d_ws,
                              size_t ws_size, hipStream_t stream) {
  (void)in_sizes; (void)n_in; (void)out_size; (void)ws_size;
  const float* x     = (const float*)d_in[0];
  const float* Wq    = (const float*)d_in[1];
  const float* bq    = (const float*)d_in[2];
  const float* Wk    = (const float*)d_in[3];
  const float* bk    = (const float*)d_in[4];
  const float* Wv    = (const float*)d_in[5];
  const float* bv    = (const float*)d_in[6];
  const float* gamma = (const float*)d_in[7];
  float* out = (float*)d_out;

  char* ws = (char*)d_ws;
  _Float16* xh = (_Float16*)(ws + 0);
  _Float16* Wh = (_Float16*)(ws + 8388608);
  _Float16* qh = (_Float16*)(ws + 8552448);
  _Float16* kh = (_Float16*)(ws + 9601024);
  _Float16* vh = (_Float16*)(ws + 10649600);

  sa_xpose_kernel<<<4096, 256, 0, stream>>>(x, xh);
  sa_wconv_kernel<<<320, 256, 0, stream>>>(Wq, Wk, Wv, Wh);
  sa_proj_kernel <<<128, 256, 0, stream>>>(xh, Wh, bq, bk, bv, qh, kh, vh);
  sa_attn_kernel <<<128, 256, 0, stream>>>(qh, kh, vh, x, gamma, out);
}
